// MultiHead_QKV_BlockAttention_42949673149
// MI455X (gfx1250) — matrix operand order repaired and checked by reading, not compiled, not run on hardware
//
#include <hip/hip_runtime.h>

#define BATCH 4
#define SEQ   2048
#define FEAT  1024
#define HEADS 16
#define HDIM  64

typedef __attribute__((ext_vector_type(16))) __bf16 v16bf;
typedef __attribute__((ext_vector_type(8)))  __bf16 v8bf;
typedef __attribute__((ext_vector_type(4)))  __bf16 v4bf;
typedef __attribute__((ext_vector_type(8)))  float  v8f;

// ---------- bf16 helpers (round-to-nearest-even) ----------
__device__ __forceinline__ float bf2f(__bf16 b) {
    unsigned short s = __builtin_bit_cast(unsigned short, b);
    unsigned int u = ((unsigned int)s) << 16;
    return __builtin_bit_cast(float, u);
}
__device__ __forceinline__ __bf16 f2bf(float x) {
    unsigned int u = __builtin_bit_cast(unsigned int, x);
    unsigned int r = (u + 0x7FFFu + ((u >> 16) & 1u)) >> 16;
    return __builtin_bit_cast(__bf16, (unsigned short)(r & 0xFFFFu));
}
struct bfpair { __bf16 hi, lo; };
__device__ __forceinline__ bfpair splitbf(float x) {
    bfpair p;
    p.hi = f2bf(x);
    p.lo = f2bf(x - bf2f(p.hi));
    return p;
}

// ---------- WMMA wrappers ----------
__device__ __forceinline__ v8f wmma_bf16(v16bf a, v16bf b, v8f c) {
    return __builtin_amdgcn_wmma_f32_16x16x32_bf16(false, a, false, b, (short)0, c,
                                                   false, false);
}
// split-bf16 emulated ~fp32 product: hi*hi + hi*lo + lo*hi
__device__ __forceinline__ v8f wmma_split(v16bf ah, v16bf al, v16bf bh, v16bf bl, v8f c) {
    c = wmma_bf16(ah, bh, c);
    c = wmma_bf16(ah, bl, c);
    c = wmma_bf16(al, bh, c);
    return c;
}

// ---------- bf16 fragment loaders from LDS (pure ds_load_b128) ----------
// A-frag 16x32: lane m = L&15 ; K(j) = ((L>=16)?16:0) + j
// Row-major bf16 source [m][k] -> one contiguous 16-elem run.
__device__ __forceinline__ v16bf load_a_frag_bf(const __bf16* base, int ld, int lane) {
    const int m  = lane & 15;
    const int kb = (lane >> 4) << 4;
    v8bf a = *(const v8bf*)(base + m * ld + kb);
    v8bf b = *(const v8bf*)(base + m * ld + kb + 8);
    return __builtin_shufflevector(a, b, 0, 1, 2, 3, 4, 5, 6, 7,
                                   8, 9, 10, 11, 12, 13, 14, 15);
}
// B-frag 32x16: lane n = L&15 ; k(j) = ((L>=16)?16:0) + j.
// Source stored K-contiguous per n:  elem = base[n*ld + kb + j]  (16 contiguous)
__device__ __forceinline__ v16bf load_bk_frag_bf(const __bf16* base, int ld, int lane) {
    const int n  = lane & 15;
    const int kb = (lane >> 4) << 4;
    const v8bf* p = (const v8bf*)(base + n * ld + kb);
    v8bf a = p[0], b = p[1];
    return __builtin_shufflevector(a, b, 0, 1, 2, 3, 4, 5, 6, 7,
                                   8, 9, 10, 11, 12, 13, 14, 15);
}
// A-frag built from fp32 global source (one-time Q load in attention kernel)
__device__ __forceinline__ void load_a_frag_f32(const float* src, int ld, int lane,
                                                v16bf& hi, v16bf& lo) {
    const int m  = lane & 15;
    const int kb = (lane >> 4) << 4;
#pragma unroll
    for (int j = 0; j < 16; ++j) {
        int k = kb + j;
        bfpair p = splitbf(src[m * ld + k]);
        hi[j] = p.hi; lo[j] = p.lo;
    }
}

// ==========================================================================
// Kernel 1: Y = X @ W  for W in {wq,wk,wv}; result scattered to [B,H,S,D] ws
// grid = (FEAT/128, (B*S)/128, 3), block = 256 (8 waves)
// ==========================================================================
__global__ __launch_bounds__(256) void qkv_proj_kernel(
    const float* __restrict__ X, const float* __restrict__ wq,
    const float* __restrict__ wk, const float* __restrict__ wv,
    float* __restrict__ ws_qkv) {
    __shared__ __align__(16) __bf16 Ahi[128][32];  // X tile, row-major [m][k]
    __shared__ __align__(16) __bf16 Alo[128][32];
    __shared__ __align__(16) __bf16 Bhi[128][32];  // W tile, transposed [n][k]
    __shared__ __align__(16) __bf16 Blo[128][32];

    const int tid   = threadIdx.x;
    const int n0    = blockIdx.x * 128;
    const int m0    = blockIdx.y * 128;
    const int which = blockIdx.z;
    const float* W  = (which == 0) ? wq : (which == 1) ? wk : wv;
    float* dst = ws_qkv + (size_t)which * ((size_t)BATCH * HEADS * SEQ * HDIM);

    const int w = tid >> 5, lane = tid & 31;
    const int wm = (w >> 1) * 32;   // 4 wave-rows of 32
    const int wn = (w & 1) * 64;    // 2 wave-cols of 64

    const v8f vzero = {0.f, 0.f, 0.f, 0.f, 0.f, 0.f, 0.f, 0.f};
    v8f acc[2][4];
#pragma unroll
    for (int i = 0; i < 2; ++i)
#pragma unroll
        for (int jn = 0; jn < 4; ++jn) acc[i][jn] = vzero;

    // W-tile staging mapping: thread owns one n-column, one 16-wide k half.
    const int bn = tid & 127;        // 0..127  (lanes cover consecutive n: coalesced)
    const int bk0 = (tid >> 7) * 16; // 0 or 16

    for (int kt = 0; kt < FEAT; kt += 32) {
        // ---- stage X tile 128x32 (row-major, convert once) ----
#pragma unroll
        for (int q = 0; q < 4; ++q) {
            int idx = tid + 256 * q;            // 1024 float4s
            int row = idx >> 3, c4 = idx & 7;
            float4 v = *reinterpret_cast<const float4*>(
                &X[(size_t)(m0 + row) * FEAT + kt + c4 * 4]);
            v4bf h, l;
            bfpair p;
            p = splitbf(v.x); h[0] = p.hi; l[0] = p.lo;
            p = splitbf(v.y); h[1] = p.hi; l[1] = p.lo;
            p = splitbf(v.z); h[2] = p.hi; l[2] = p.lo;
            p = splitbf(v.w); h[3] = p.hi; l[3] = p.lo;
            *reinterpret_cast<v4bf*>(&Ahi[row][c4 * 4]) = h;
            *reinterpret_cast<v4bf*>(&Alo[row][c4 * 4]) = l;
        }
        // ---- stage W tile 32x128 transposed -> [n][k] ----
        {
            float col[16];
#pragma unroll
            for (int kk = 0; kk < 16; ++kk)
                col[kk] = W[(size_t)(kt + bk0 + kk) * FEAT + n0 + bn];
            v8bf h0, l0, h1, l1;
#pragma unroll
            for (int kk = 0; kk < 8; ++kk) {
                bfpair p0 = splitbf(col[kk]);
                bfpair p1 = splitbf(col[8 + kk]);
                h0[kk] = p0.hi; l0[kk] = p0.lo;
                h1[kk] = p1.hi; l1[kk] = p1.lo;
            }
            *reinterpret_cast<v8bf*>(&Bhi[bn][bk0])     = h0;
            *reinterpret_cast<v8bf*>(&Bhi[bn][bk0 + 8]) = h1;
            *reinterpret_cast<v8bf*>(&Blo[bn][bk0])     = l0;
            *reinterpret_cast<v8bf*>(&Blo[bn][bk0 + 8]) = l1;
        }
        __syncthreads();

        v16bf ah[2], al[2], bh[4], bl[4];
#pragma unroll
        for (int i = 0; i < 2; ++i) {
            ah[i] = load_a_frag_bf(&Ahi[wm + i * 16][0], 32, lane);
            al[i] = load_a_frag_bf(&Alo[wm + i * 16][0], 32, lane);
        }
#pragma unroll
        for (int jn = 0; jn < 4; ++jn) {
            bh[jn] = load_bk_frag_bf(&Bhi[wn + jn * 16][0], 32, lane);
            bl[jn] = load_bk_frag_bf(&Blo[wn + jn * 16][0], 32, lane);
        }
#pragma unroll
        for (int i = 0; i < 2; ++i)
#pragma unroll
            for (int jn = 0; jn < 4; ++jn)
                acc[i][jn] = wmma_split(ah[i], al[i], bh[jn], bl[jn], acc[i][jn]);
        __syncthreads();
    }

    // scatter C tiles (16x16 f32 layout: row = r + (L>=16?8:0), col = L&15)
    const int hi8 = (lane >> 4) << 3;
    const int nn  = lane & 15;
#pragma unroll
    for (int i = 0; i < 2; ++i)
#pragma unroll
        for (int jn = 0; jn < 4; ++jn)
#pragma unroll
            for (int r = 0; r < 8; ++r) {
                int m = m0 + wm + i * 16 + r + hi8;   // row in [B*S]
                int n = n0 + wn + jn * 16 + nn;       // col in [F]
                int b = m >> 11, s = m & (SEQ - 1);
                int h = n >> 6,  d = n & (HDIM - 1);
                dst[(((size_t)(b * HEADS + h)) * SEQ + s) * HDIM + d] = acc[i][jn][r];
            }
}

// ==========================================================================
// Kernel 2: flash attention over ws Q/K/V [B,H,S,D] + residual -> out (twice)
// grid = (SEQ/64, B*H), block = 128 (4 waves, 16 query rows each)
// ==========================================================================
__global__ __launch_bounds__(128) void flash_attn_kernel(
    const float* __restrict__ ws_qkv, const float* __restrict__ residual,
    float* __restrict__ out) {
    __shared__ __align__(16) __bf16 Khi[32][HDIM];   // row-major [t][d]
    __shared__ __align__(16) __bf16 Klo[32][HDIM];
    __shared__ __align__(16) __bf16 Vthi[HDIM][32];  // transposed [d][t]
    __shared__ __align__(16) __bf16 Vtlo[HDIM][32];
    __shared__ __align__(16) __bf16 Phi[4][16][32];  // per-wave P, row-major
    __shared__ __align__(16) __bf16 Plo[4][16][32];

    const int tid  = threadIdx.x;
    const int w    = tid >> 5, lane = tid & 31;
    const int bh   = blockIdx.y;
    const int b    = bh >> 4, h = bh & 15;
    const int q0   = blockIdx.x * 64 + w * 16;   // this wave's first query row

    const size_t plane = (size_t)SEQ * HDIM;
    const size_t mats  = (size_t)BATCH * HEADS * plane;
    const float* Qp = ws_qkv + (size_t)bh * plane;
    const float* Kp = ws_qkv + mats + (size_t)bh * plane;
    const float* Vp = ws_qkv + 2 * mats + (size_t)bh * plane;

    const float invscale = 0.08838834764831843f;   // 1/sqrt(S/H) = 1/sqrt(128)

    // Q A-frags for d0 = 0, 32 (reused across all key tiles)
    v16bf qah[2], qal[2];
#pragma unroll
    for (int kk = 0; kk < 2; ++kk)
        load_a_frag_f32(Qp + (size_t)q0 * HDIM + kk * 32, HDIM, lane, qah[kk], qal[kk]);

    const v8f vzero = {0.f, 0.f, 0.f, 0.f, 0.f, 0.f, 0.f, 0.f};
    v8f o[4];
#pragma unroll
    for (int dt = 0; dt < 4; ++dt) o[dt] = vzero;
    float mrow[8], lrow[8];
#pragma unroll
    for (int r = 0; r < 8; ++r) { mrow[r] = -3.0e38f; lrow[r] = 0.f; }

    // V-tile transposed staging mapping: thread owns one d-row, one 16-wide t half
    const int vd  = tid & 63;        // lanes cover consecutive d: coalesced
    const int vt0 = (tid >> 6) * 16; // 0 or 16

    for (int t0 = 0; t0 < SEQ; t0 += 32) {
        // ---- stage K tile 32x64 row-major (convert once) ----
#pragma unroll
        for (int q = 0; q < 4; ++q) {
            int idx = tid + 128 * q;           // 512 float4s
            int row = idx >> 4, c4 = idx & 15;
            float4 v = *reinterpret_cast<const float4*>(
                &Kp[(size_t)(t0 + row) * HDIM + c4 * 4]);
            v4bf hh, ll;
            bfpair p;
            p = splitbf(v.x); hh[0] = p.hi; ll[0] = p.lo;
            p = splitbf(v.y); hh[1] = p.hi; ll[1] = p.lo;
            p = splitbf(v.z); hh[2] = p.hi; ll[2] = p.lo;
            p = splitbf(v.w); hh[3] = p.hi; ll[3] = p.lo;
            *reinterpret_cast<v4bf*>(&Khi[row][c4 * 4]) = hh;
            *reinterpret_cast<v4bf*>(&Klo[row][c4 * 4]) = ll;
        }
        // ---- stage V tile 32x64 transposed -> [d][t] ----
        {
            float col[16];
#pragma unroll
            for (int tt = 0; tt < 16; ++tt)
                col[tt] = Vp[(size_t)(t0 + vt0 + tt) * HDIM + vd];
            v8bf h0, l0, h1, l1;
#pragma unroll
            for (int tt = 0; tt < 8; ++tt) {
                bfpair p0 = splitbf(col[tt]);
                bfpair p1 = splitbf(col[8 + tt]);
                h0[tt] = p0.hi; l0[tt] = p0.lo;
                h1[tt] = p1.hi; l1[tt] = p1.lo;
            }
            *reinterpret_cast<v8bf*>(&Vthi[vd][vt0])     = h0;
            *reinterpret_cast<v8bf*>(&Vthi[vd][vt0 + 8]) = h1;
            *reinterpret_cast<v8bf*>(&Vtlo[vd][vt0])     = l0;
            *reinterpret_cast<v8bf*>(&Vtlo[vd][vt0 + 8]) = l1;
        }
        __syncthreads();

        // S = Q . K^T  (two 16x16 logit tiles, reduce over d = 64)
        // B[k=d][n=t] = K[t][d]: d is contiguous in K rows -> load_bk works directly
        v8f s0 = vzero, s1 = vzero;
#pragma unroll
        for (int kk = 0; kk < 2; ++kk) {
            v16bf bhh, bll;
            bhh = load_bk_frag_bf(&Khi[0][kk * 32], HDIM, lane);
            bll = load_bk_frag_bf(&Klo[0][kk * 32], HDIM, lane);
            s0 = wmma_split(qah[kk], qal[kk], bhh, bll, s0);
            bhh = load_bk_frag_bf(&Khi[16][kk * 32], HDIM, lane);
            bll = load_bk_frag_bf(&Klo[16][kk * 32], HDIM, lane);
            s1 = wmma_split(qah[kk], qal[kk], bhh, bll, s1);
        }

        // online softmax (each lane owns 8 rows; a row spans 16 lanes)
#pragma unroll
        for (int r = 0; r < 8; ++r) {
            float a = s0[r] * invscale;
            float c = s1[r] * invscale;
            float rm = fmaxf(a, c);
            rm = fmaxf(rm, __shfl_xor(rm, 1));
            rm = fmaxf(rm, __shfl_xor(rm, 2));
            rm = fmaxf(rm, __shfl_xor(rm, 4));
            rm = fmaxf(rm, __shfl_xor(rm, 8));
            float mnew  = fmaxf(mrow[r], rm);
            float alpha = __expf(mrow[r] - mnew);
            float p0 = __expf(a - mnew);
            float p1 = __expf(c - mnew);
            float rs = p0 + p1;
            rs += __shfl_xor(rs, 1);
            rs += __shfl_xor(rs, 2);
            rs += __shfl_xor(rs, 4);
            rs += __shfl_xor(rs, 8);
            lrow[r] = lrow[r] * alpha + rs;
            mrow[r] = mnew;
#pragma unroll
            for (int dt = 0; dt < 4; ++dt) o[dt][r] *= alpha;
            int row = r + ((lane >> 4) << 3);
            bfpair pp;
            pp = splitbf(p0);
            Phi[w][row][lane & 15] = pp.hi;
            Plo[w][row][lane & 15] = pp.lo;
            pp = splitbf(p1);
            Phi[w][row][(lane & 15) + 16] = pp.hi;
            Plo[w][row][(lane & 15) + 16] = pp.lo;
        }

        // O += P(16x32) @ V(32x64)  — same-wave LDS RAW is in-order
        v16bf pah = load_a_frag_bf(&Phi[w][0][0], 32, lane);
        v16bf pal = load_a_frag_bf(&Plo[w][0][0], 32, lane);
#pragma unroll
        for (int dt = 0; dt < 4; ++dt) {
            v16bf vbh = load_bk_frag_bf(&Vthi[dt * 16][0], 32, lane);
            v16bf vbl = load_bk_frag_bf(&Vtlo[dt * 16][0], 32, lane);
            o[dt] = wmma_split(pah, pal, vbh, vbl, o[dt]);
        }
        __syncthreads();
    }

    // epilogue: normalize, add residual, write both tuple halves
    const int hi8 = (lane >> 4) << 3;
    const size_t HALF = (size_t)BATCH * SEQ * FEAT;
#pragma unroll
    for (int dt = 0; dt < 4; ++dt)
#pragma unroll
        for (int r = 0; r < 8; ++r) {
            int srow = q0 + r + hi8;
            int f    = h * HDIM + dt * 16 + (lane & 15);
            size_t idx = ((size_t)b * SEQ + srow) * FEAT + f;
            float val = o[dt][r] / lrow[r] + residual[idx];
            out[idx]        = val;
            out[HALF + idx] = val;
        }
}

// ==========================================================================
extern "C" void kernel_launch(void* const* d_in, const int* in_sizes, int n_in,
                              void* d_out, int out_size, void* d_ws, size_t ws_size,
                              hipStream_t stream) {
    const float* X        = (const float*)d_in[0];
    const float* residual = (const float*)d_in[1];
    const float* wq       = (const float*)d_in[2];
    const float* wk       = (const float*)d_in[3];
    const float* wv       = (const float*)d_in[4];
    float* out = (float*)d_out;
    float* ws  = (float*)d_ws;   // needs 3 * B*H*S*D floats = ~100.7 MB

    dim3 gridA(FEAT / 128, (BATCH * SEQ) / 128, 3);
    qkv_proj_kernel<<<gridA, 256, 0, stream>>>(X, wq, wk, wv, ws);

    dim3 gridB(SEQ / 64, BATCH * HEADS, 1);
    flash_attn_kernel<<<gridB, 128, 0, stream>>>(ws, residual, out);
}
